// Attention_22892175688076
// MI455X (gfx1250) — matrix operand order repaired and checked by reading, not compiled, not run on hardware
//
#include <hip/hip_runtime.h>

// Problem constants (fixed by the reference)
#define B_DIM 2
#define S_LEN 2048
#define E_DIM 4096
#define NH_ 32
#define NKV_ 8
#define HD_ 128

typedef __attribute__((ext_vector_type(16))) __bf16 bf16x16;
typedef __attribute__((ext_vector_type(8)))  __bf16 bf16x8;
typedef __attribute__((ext_vector_type(4)))  __bf16 bf16x4;
typedef __attribute__((ext_vector_type(8)))  float  v8f;
typedef __attribute__((ext_vector_type(4)))  float  f32x4;

// Detect the gfx1250 async global->LDS builtins; fall back to sync loads.
#if defined(__has_builtin)
#if __has_builtin(__builtin_amdgcn_global_load_async_to_lds_b128) && \
    __has_builtin(__builtin_amdgcn_s_wait_asynccnt)
#define FA_ASYNC 1
#endif
#endif
#ifndef FA_ASYNC
#define FA_ASYNC 0
#endif

#if FA_ASYNC
// Builtin signature (from hipcc diagnostic): param0 is
//   '__attribute__((__vector_size__(4*sizeof(int)))) int __device__ *'
// i.e. pointer to a GCC int vector of 4 in the global AS; dst is the LDS AS.
typedef int i32x4_gcc __attribute__((vector_size(16)));
typedef __attribute__((address_space(1))) i32x4_gcc as1_i32x4;
typedef __attribute__((address_space(3))) i32x4_gcc as3_i32x4;
__device__ __forceinline__ void g2l_async_b128(const void* g, const void* l) {
  // generic->AS pointer via integer detour: global generic addr == global addr,
  // LDS generic addr low 32 bits == LDS offset (aperture mapping).
  __builtin_amdgcn_global_load_async_to_lds_b128(
      (as1_i32x4*)(unsigned long long)g,
      (as3_i32x4*)(unsigned int)(unsigned long long)l,
      /*offset=*/0, /*cpol=*/0);
}
#endif

// ---------------------------------------------------------------------------
// WMMA helpers (CDNA5 wave32, v_wmma_f32_16x16x32_bf16)
// ---------------------------------------------------------------------------
__device__ __forceinline__ v8f wmma_bf16(bf16x16 a, bf16x16 b, v8f c) {
  return __builtin_amdgcn_wmma_f32_16x16x32_bf16(
      /*neg_a=*/false, a, /*neg_b=*/false, b,
      /*c_mod=*/(short)0, c, /*reuse_a=*/false, /*reuse_b=*/false);
}

// A-matrix 16x32 bf16 fragment from a row-major [16][>=32] tile (stride ld).
// lanes 0-15: elems 0-7 = K 0..7,  elems 8-15 = K 16..23  (row = lane)
// lanes16-31: elems 0-7 = K 8..15, elems 8-15 = K 24..31  (row = lane-16)
__device__ __forceinline__ bf16x16 load_a_frag(const __bf16* base, int ld) {
  const int lane = threadIdx.x & 31;
  const int row  = lane & 15;
  const int hi   = lane >> 4;
  const __bf16* p = base + row * ld + hi * 8;
  bf16x8 lo = *(const bf16x8*)(p);
  bf16x8 h8 = *(const bf16x8*)(p + 16);
  bf16x16 r;
#pragma unroll
  for (int i = 0; i < 8; ++i) { r[i] = lo[i]; r[8 + i] = h8[i]; }
  return r;
}

// B-matrix 32x16 bf16 fragment from a K-transposed tile Bt[n][k] (stride ld).
// lanes 0-15: elems 0-7 = K 0..7, 8-15 = K 16..23 ; lanes 16-31: elems 0-7 = K 8..15, 8-15 = K 24..31 (col = lane&15)
__device__ __forceinline__ bf16x16 load_b_frag(const __bf16* base, int ld) {
  const int lane = threadIdx.x & 31;
  const int col  = lane & 15;
  const int hi   = lane >> 4;
  const __bf16* p = base + col * ld + hi * 8;
  bf16x8 lo = *(const bf16x8*)(p);
  bf16x8 h8 = *(const bf16x8*)(p + 16);
  bf16x16 r;
#pragma unroll
  for (int i = 0; i < 8; ++i) { r[i] = lo[i]; r[8 + i] = h8[i]; }
  return r;
}

// ---------------------------------------------------------------------------
// fp32 -> bf16 conversion (all sizes here are multiples of 1024)
// ---------------------------------------------------------------------------
__global__ __launch_bounds__(256) void cvt_f32_to_bf16(
    const float* __restrict__ x, __bf16* __restrict__ y, long long n) {
  long long i = ((long long)blockIdx.x * 256 + threadIdx.x) * 4;
  if (i + 3 < n) {
    f32x4 v = *(const f32x4*)(x + i);
    bf16x4 o;
    o[0] = (__bf16)v[0]; o[1] = (__bf16)v[1];
    o[2] = (__bf16)v[2]; o[3] = (__bf16)v[3];
    *(bf16x4*)(y + i) = o;
  }
}

// ---------------------------------------------------------------------------
// bf16 GEMM, C = A[M,K] * B[K,N].  Tile 128x128x32, 256 thr = 8 waves (2x4),
// each wave computes a 64x32 sub-tile (4 x 2 WMMA accumulators).
// Register-staged pipeline: next tile's global loads issue before the WMMA
// phase; the loadcnt wait lands at the LDS store after the barrier.
// MODE 0: fp32 output, plain row-major [M,N]
// MODE 1: bf16 output, head-major  [(b*NHEADS+h), s, d]  (m=b*S+s, n=h*HD+d)
// ---------------------------------------------------------------------------
template <int MODE, int NHEADS>
__global__ __launch_bounds__(256) void gemm_bf16_128x128(
    const __bf16* __restrict__ A, const __bf16* __restrict__ Bm,
    float* __restrict__ Cf, __bf16* __restrict__ Cb,
    int M, int N, int K) {
  __shared__ __bf16 As[128 * 40];  // [row][k], padded stride 40
  __shared__ __bf16 Bs[128 * 40];  // transposed [n][k], padded stride 40

  const int tid    = threadIdx.x;
  const int wave   = tid >> 5;
  const int lane   = tid & 31;
  const int ln     = lane & 15;
  const int hi     = lane >> 4;
  const int wave_m = wave >> 2;  // 0..1
  const int wave_n = wave & 3;   // 0..3
  const int m0 = blockIdx.y * 128;
  const int n0 = blockIdx.x * 128;

  // Per-thread tile-load coordinates (loop invariant)
  int a_r[4], a_c[4], b_k[4], b_n[4];
#pragma unroll
  for (int i = 0; i < 4; ++i) {
    int v = tid + i * 256;       // 0..1023
    a_r[i] = v >> 3;             // 0..127
    a_c[i] = (v & 7) << 2;       // 0..28
    b_k[i] = v >> 5;             // 0..31
    b_n[i] = (v & 31) << 2;      // 0..124
  }

  v8f acc[4][2];
#pragma unroll
  for (int mi = 0; mi < 4; ++mi)
#pragma unroll
    for (int ni = 0; ni < 2; ++ni)
#pragma unroll
      for (int e = 0; e < 8; ++e) acc[mi][ni][e] = 0.0f;

  bf16x4 ra[4], rb[4];
  // Prologue: tile 0 -> regs -> LDS
#pragma unroll
  for (int i = 0; i < 4; ++i) {
    ra[i] = *(const bf16x4*)(A + (size_t)(m0 + a_r[i]) * K + a_c[i]);
    rb[i] = *(const bf16x4*)(Bm + (size_t)b_k[i] * N + n0 + b_n[i]);
  }
#pragma unroll
  for (int i = 0; i < 4; ++i) {
    *(bf16x4*)(&As[a_r[i] * 40 + a_c[i]]) = ra[i];
#pragma unroll
    for (int j = 0; j < 4; ++j) Bs[(b_n[i] + j) * 40 + b_k[i]] = rb[i][j];
  }
  __syncthreads();

  int k0 = 0;
  while (true) {
    const bool has_next = (k0 + 32 < K);
    if (has_next) {  // issue next tile's global loads; consumed post-barrier
      const int kn = k0 + 32;
#pragma unroll
      for (int i = 0; i < 4; ++i) {
        ra[i] = *(const bf16x4*)(A + (size_t)(m0 + a_r[i]) * K + kn + a_c[i]);
        rb[i] = *(const bf16x4*)(Bm + (size_t)(kn + b_k[i]) * N + n0 + b_n[i]);
      }
    }
    // Batch all fragment loads, then run the 8-WMMA block
    bf16x16 af[4], bfr[2];
#pragma unroll
    for (int mi = 0; mi < 4; ++mi)
      af[mi] = load_a_frag(&As[(wave_m * 64 + mi * 16) * 40], 40);
#pragma unroll
    for (int ni = 0; ni < 2; ++ni)
      bfr[ni] = load_b_frag(&Bs[(wave_n * 32 + ni * 16) * 40], 40);
#pragma unroll
    for (int mi = 0; mi < 4; ++mi)
#pragma unroll
      for (int ni = 0; ni < 2; ++ni)
        acc[mi][ni] = wmma_bf16(af[mi], bfr[ni], acc[mi][ni]);

    if (!has_next) break;
    __syncthreads();
#pragma unroll
    for (int i = 0; i < 4; ++i) {
      *(bf16x4*)(&As[a_r[i] * 40 + a_c[i]]) = ra[i];
#pragma unroll
      for (int j = 0; j < 4; ++j) Bs[(b_n[i] + j) * 40 + b_k[i]] = rb[i][j];
    }
    __syncthreads();
    k0 += 32;
  }

  // Epilogue: C layout -> row = local 16-tile row (hi*8+e), col = lane&15
#pragma unroll
  for (int mi = 0; mi < 4; ++mi)
#pragma unroll
    for (int ni = 0; ni < 2; ++ni)
#pragma unroll
      for (int e = 0; e < 8; ++e) {
        int m = m0 + wave_m * 64 + mi * 16 + hi * 8 + e;
        int n = n0 + wave_n * 32 + ni * 16 + ln;
        float val = acc[mi][ni][e];
        if (MODE == 0) {
          Cf[(size_t)m * N + n] = val;
        } else {
          int b = m >> 11, s = m & (S_LEN - 1);   // S = 2048
          int h = n >> 7,  d = n & (HD_ - 1);     // HD = 128
          Cb[(((size_t)(b * NHEADS + h) * S_LEN + s) << 7) + d] = (__bf16)val;
        }
      }
}

// ---------------------------------------------------------------------------
// Flash attention (causal, online softmax).  Grid: (S/64, B*NH), 128 threads.
// Q,K,V bf16 head-major [B*H, S, HD].  Each wave owns 16 query rows.
// K tile copied with async global->LDS (overlapped with the V transpose);
// output bf16 [B*S, NH*HD] for the final projection GEMM.
// ---------------------------------------------------------------------------
__global__ __launch_bounds__(128) void flash_attn_causal(
    const __bf16* __restrict__ Q, const __bf16* __restrict__ Km,
    const __bf16* __restrict__ Vm, __bf16* __restrict__ O) {
  __shared__ __bf16 Kt[64 * 136];      // [key][hd]  (natural = B^T for QK^T)
  __shared__ __bf16 Vt[128 * 72];      // [hd][key]  (transposed for P*V)
  __shared__ __bf16 Pb[4][16 * 72];    // per-wave P staging, [row][key]

  const int tid  = threadIdx.x;
  const int wave = tid >> 5;
  const int lane = tid & 31;
  const int ln   = lane & 15;
  const int hi   = lane >> 4;
  const int qb   = blockIdx.x;               // query tile (64 rows)
  const int bh   = blockIdx.y;               // b*NH + h
  const int b    = bh >> 5;
  const int h    = bh & 31;
  const int kvh  = h >> 2;                   // N_REP = 4

  const __bf16* Qg = Q  + (size_t)bh * S_LEN * HD_;
  const __bf16* Kg = Km + (size_t)(b * NKV_ + kvh) * S_LEN * HD_;
  const __bf16* Vg = Vm + (size_t)(b * NKV_ + kvh) * S_LEN * HD_;

  // Resident Q fragments: 4 chunks of K=32 over HD=128
  bf16x16 qf[4];
  {
    const int qrow = qb * 64 + wave * 16 + ln;
    const __bf16* qp = Qg + (size_t)qrow * HD_;
#pragma unroll
    for (int c = 0; c < 4; ++c) {
      bf16x8 lo = *(const bf16x8*)(qp + c * 32 + hi * 8);
      bf16x8 h8 = *(const bf16x8*)(qp + c * 32 + hi * 8 + 16);
#pragma unroll
      for (int i = 0; i < 8; ++i) { qf[c][i] = lo[i]; qf[c][8 + i] = h8[i]; }
    }
  }

  v8f acc[8];
#pragma unroll
  for (int nt = 0; nt < 8; ++nt)
#pragma unroll
    for (int e = 0; e < 8; ++e) acc[nt][e] = 0.0f;

  float mrow[8], lrow[8];
#pragma unroll
  for (int e = 0; e < 8; ++e) { mrow[e] = -1e30f; lrow[e] = 0.0f; }

  const float qk_scale = 0.08838834764831845f;  // 1/sqrt(128)

  for (int t = 0; t <= qb; ++t) {
    const int kbase = t * 64;
    // --- K tile: 64 keys x 128 hd, natural layout -----------------------
#if FA_ASYNC
#pragma unroll
    for (int i = 0; i < 8; ++i) {
      int v  = tid + i * 128;            // 0..1023 b128 chunks
      int r  = v >> 4;                   // key row 0..63
      int c8 = (v & 15) << 3;            // hd 0..120
      g2l_async_b128(Kg + (size_t)(kbase + r) * HD_ + c8, &Kt[r * 136 + c8]);
    }
#else
#pragma unroll
    for (int i = 0; i < 8; ++i) {
      int v  = tid + i * 128;
      int r  = v >> 4;
      int c8 = (v & 15) << 3;
      bf16x8 dk = *(const bf16x8*)(Kg + (size_t)(kbase + r) * HD_ + c8);
      *(bf16x8*)(&Kt[r * 136 + c8]) = dk;
    }
#endif
    // --- V tile: transpose into LDS via VGPRs (overlaps async K copy) ---
#pragma unroll
    for (int i = 0; i < 8; ++i) {
      int v  = tid + i * 128;
      int r  = v >> 4;
      int c8 = (v & 15) << 3;
      bf16x8 dv = *(const bf16x8*)(Vg + (size_t)(kbase + r) * HD_ + c8);
#pragma unroll
      for (int j = 0; j < 8; ++j) Vt[(c8 + j) * 72 + r] = dv[j];
    }
    if (t < qb) {  // prefetch next tile
      int r = tid >> 1;
      const __bf16* np = ((tid & 1) ? Vg : Kg) + (size_t)(kbase + 64 + r) * HD_;
      __builtin_prefetch(np, 0, 0);
    }
#if FA_ASYNC
    __builtin_amdgcn_s_wait_asynccnt(0);
#endif
    __syncthreads();

    // --- Scores: 4 sub-tiles of 16 keys; batch loads, then WMMA chain ---
    float s4[4][8];
#pragma unroll
    for (int st = 0; st < 4; ++st) {
      bf16x16 kf[4];
#pragma unroll
      for (int c = 0; c < 4; ++c)
        kf[c] = load_b_frag(&Kt[st * 16 * 136 + c * 32], 136);
      v8f s;
#pragma unroll
      for (int e = 0; e < 8; ++e) s[e] = 0.0f;
#pragma unroll
      for (int c = 0; c < 4; ++c) s = wmma_bf16(qf[c], kf[c], s);
#pragma unroll
      for (int e = 0; e < 8; ++e) {
        float x = s[e] * qk_scale;
        if (t == qb) {  // causal mask on the diagonal tile
          int ql = wave * 16 + hi * 8 + e;
          int kl = st * 16 + ln;
          if (kl > ql) x = -1e30f;
        }
        s4[st][e] = x;
      }
    }

    // --- Online softmax over the 64 keys --------------------------------
    float mt[8];
#pragma unroll
    for (int e = 0; e < 8; ++e) {
      float v = fmaxf(fmaxf(s4[0][e], s4[1][e]), fmaxf(s4[2][e], s4[3][e]));
#pragma unroll
      for (int msk = 1; msk <= 8; msk <<= 1) v = fmaxf(v, __shfl_xor(v, msk));
      mt[e] = v;
    }
    float mn[8], sca[8], rs[8];
#pragma unroll
    for (int e = 0; e < 8; ++e) {
      mn[e]  = fmaxf(mrow[e], mt[e]);
      sca[e] = __expf(mrow[e] - mn[e]);
      rs[e]  = 0.0f;
    }
#pragma unroll
    for (int st = 0; st < 4; ++st)
#pragma unroll
      for (int e = 0; e < 8; ++e) {
        float p = __expf(s4[st][e] - mn[e]);
        rs[e] += p;
        Pb[wave][(hi * 8 + e) * 72 + st * 16 + ln] = (__bf16)p;
      }
#pragma unroll
    for (int e = 0; e < 8; ++e) {
#pragma unroll
      for (int msk = 1; msk <= 8; msk <<= 1) rs[e] += __shfl_xor(rs[e], msk);
      lrow[e] = lrow[e] * sca[e] + rs[e];
      mrow[e] = mn[e];
    }
#pragma unroll
    for (int nt = 0; nt < 8; ++nt)
#pragma unroll
      for (int e = 0; e < 8; ++e) acc[nt][e] *= sca[e];

    // --- O += P(16x64) * V(64x128): batch V frags in groups of 4 --------
#pragma unroll
    for (int c2 = 0; c2 < 2; ++c2) {
      bf16x16 pf = load_a_frag(&Pb[wave][c2 * 32], 72);
#pragma unroll
      for (int g = 0; g < 2; ++g) {
        bf16x16 vf[4];
#pragma unroll
        for (int i = 0; i < 4; ++i)
          vf[i] = load_b_frag(&Vt[(g * 4 + i) * 16 * 72 + c2 * 32], 72);
#pragma unroll
        for (int i = 0; i < 4; ++i)
          acc[g * 4 + i] = wmma_bf16(pf, vf[i], acc[g * 4 + i]);
      }
    }
    __syncthreads();
  }

  // Normalize and store bf16 to [B*S, NH*HD]
  float inv[8];
#pragma unroll
  for (int e = 0; e < 8; ++e) inv[e] = 1.0f / lrow[e];
#pragma unroll
  for (int e = 0; e < 8; ++e) {
    int q = qb * 64 + wave * 16 + hi * 8 + e;
    size_t rowbase = ((size_t)(b * S_LEN + q) * (NH_ * HD_)) + (size_t)h * HD_;
#pragma unroll
    for (int nt = 0; nt < 8; ++nt)
      O[rowbase + nt * 16 + ln] = (__bf16)(acc[nt][e] * inv[e]);
  }
}

// ---------------------------------------------------------------------------
// Launch
// ---------------------------------------------------------------------------
extern "C" void kernel_launch(void* const* d_in, const int* in_sizes, int n_in,
                              void* d_out, int out_size, void* d_ws, size_t ws_size,
                              hipStream_t stream) {
  const float* hs = (const float*)d_in[0];
  const float* Wq = (const float*)d_in[1];
  const float* Wk = (const float*)d_in[2];
  const float* Wv = (const float*)d_in[3];
  const float* Wo = (const float*)d_in[4];
  float* out = (float*)d_out;

  const long long HS  = (long long)B_DIM * S_LEN * E_DIM;   // 16,777,216
  const long long WQN = (long long)E_DIM * NH_ * HD_;       // 16,777,216
  const long long WKN = (long long)E_DIM * NKV_ * HD_;      //  4,194,304

  size_t off = 0;
  __bf16* base = (__bf16*)d_ws;
  auto carve = [&](long long n) { __bf16* p = base + off; off += (size_t)n; return p; };
  __bf16* hs_b = carve(HS);
  __bf16* wq_b = carve(WQN);
  __bf16* wk_b = carve(WKN);
  __bf16* wv_b = carve(WKN);
  __bf16* wo_b = carve(WQN);
  __bf16* q_b  = carve(HS);    // [B*NH, S, HD]
  __bf16* k_b  = carve(WKN);   // [B*NKV, S, HD]
  __bf16* v_b  = carve(WKN);   // [B*NKV, S, HD]
  __bf16* ao_b = carve(HS);    // [B*S, NH*HD]

  // 1) fp32 -> bf16
  cvt_f32_to_bf16<<<(unsigned)(HS  / 1024), 256, 0, stream>>>(hs, hs_b, HS);
  cvt_f32_to_bf16<<<(unsigned)(WQN / 1024), 256, 0, stream>>>(Wq, wq_b, WQN);
  cvt_f32_to_bf16<<<(unsigned)(WKN / 1024), 256, 0, stream>>>(Wk, wk_b, WKN);
  cvt_f32_to_bf16<<<(unsigned)(WKN / 1024), 256, 0, stream>>>(Wv, wv_b, WKN);
  cvt_f32_to_bf16<<<(unsigned)(WQN / 1024), 256, 0, stream>>>(Wo, wo_b, WQN);

  const int M = B_DIM * S_LEN;  // 4096
  // 2) QKV projections (head-major bf16 output)
  gemm_bf16_128x128<1, NH_><<<dim3(E_DIM / 128, M / 128), 256, 0, stream>>>(
      hs_b, wq_b, nullptr, q_b, M, E_DIM, E_DIM);
  gemm_bf16_128x128<1, NKV_><<<dim3((NKV_ * HD_) / 128, M / 128), 256, 0, stream>>>(
      hs_b, wk_b, nullptr, k_b, M, NKV_ * HD_, E_DIM);
  gemm_bf16_128x128<1, NKV_><<<dim3((NKV_ * HD_) / 128, M / 128), 256, 0, stream>>>(
      hs_b, wv_b, nullptr, v_b, M, NKV_ * HD_, E_DIM);

  // 3) causal flash attention
  flash_attn_causal<<<dim3(S_LEN / 64, B_DIM * NH_), 128, 0, stream>>>(
      q_b, k_b, v_b, ao_b);

  // 4) output projection -> fp32 d_out
  gemm_bf16_128x128<0, 0><<<dim3(E_DIM / 128, M / 128), 256, 0, stream>>>(
      ao_b, wo_b, out, nullptr, M, E_DIM, E_DIM);
}
